// Attention_10222022164850
// MI455X (gfx1250) — matrix operand order repaired and checked by reading, not compiled, not run on hardware
//
#include <hip/hip_runtime.h>

#define B_    8
#define C_    512
#define L_    2048
#define H_    8
#define DH_   64
#define HID_  512

typedef __attribute__((ext_vector_type(16))) __bf16       v16bf;
typedef __attribute__((ext_vector_type(8)))  float        v8f;
typedef __attribute__((ext_vector_type(4)))  unsigned int u32x4;

union Frag16 {
  v16bf v;
  unsigned int u[8];
  unsigned short h[16];
};

// Native conversion: clang lowers float->__bf16 (RNE) to the hw cvt op.
__device__ __forceinline__ unsigned short f2bf(float f) {
  union { __bf16 h; unsigned short u; } cv;
  cv.h = (__bf16)f;
  return cv.u;
}
__device__ __forceinline__ unsigned int pack2bf(float a, float b) {
  union { __bf16 h[2]; unsigned int u; } cv;
  cv.h[0] = (__bf16)a; cv.h[1] = (__bf16)b;
  return cv.u;
}

// --- CDNA5 async global->LDS copy (ASYNCcnt path), with safe fallback -------
#if __has_builtin(__builtin_amdgcn_global_load_async_to_lds_b128)
#define USE_ASYNC 1
#endif

typedef int v4i __attribute__((__vector_size__(16)));
typedef __attribute__((address_space(1))) v4i v4i_g;   // global
typedef __attribute__((address_space(3))) v4i v4i_l;   // LDS

__device__ __forceinline__ void async_copy16(const unsigned short* g,
                                             unsigned short* l) {
#ifdef USE_ASYNC
  __builtin_amdgcn_global_load_async_to_lds_b128(
      (v4i_g*)(void*)g, (v4i_l*)(void*)l, 0, 0);
#else
  *(u32x4*)l = *(const u32x4*)g;   // sync fallback: load + ds_store
#endif
}
__device__ __forceinline__ void wait_async_le(int n) {
#ifdef USE_ASYNC
  if (n == 0) asm volatile("s_wait_asynccnt 0x0" ::: "memory");
  else        asm volatile("s_wait_asynccnt 0x2" ::: "memory");
#else
  (void)n;
#endif
}

// ---------------------------------------------------------------------------
// Prep 1: fp32 weights -> bf16 (pairwise, b64 in / b32 out)
// ---------------------------------------------------------------------------
__global__ __launch_bounds__(256)
void cvt_w_kernel(const float* __restrict__ src, unsigned int* __restrict__ dst,
                  int npairs) {
  int i = blockIdx.x * 256 + threadIdx.x;
  if (i < npairs) {
    float2 f = ((const float2*)src)[i];
    dst[i] = pack2bf(f.x, f.y);
  }
}

// ---------------------------------------------------------------------------
// Prep 2: x[b,c,l] fp32 -> xt[b,l,c] bf16 via LDS 32x33 tile transpose
// ---------------------------------------------------------------------------
__global__ __launch_bounds__(256)
void xpose_kernel(const float* __restrict__ x, unsigned short* __restrict__ xt) {
  __shared__ float tile[32][33];
  const int b  = blockIdx.z;
  const int c0 = blockIdx.y * 32, l0 = blockIdx.x * 32;
  const int tx = threadIdx.x & 31, ty = threadIdx.x >> 5;   // ty: 0..7
#pragma unroll
  for (int i = 0; i < 4; ++i) {
    int c = c0 + ty + 8 * i;
    tile[ty + 8 * i][tx] = x[((size_t)b * C_ + c) * L_ + l0 + tx];
  }
  __syncthreads();
#pragma unroll
  for (int i = 0; i < 4; ++i) {
    int l = l0 + ty + 8 * i;
    xt[((size_t)b * L_ + l) * C_ + c0 + tx] = f2bf(tile[tx][ty + 8 * i]);
  }
}

// A-fragment: per-lane 16 bf16 = two contiguous 8-element runs (b128 each).
__device__ __forceinline__ Frag16 load_afrag(const unsigned short* base) {
  Frag16 a;
  u32x4 a0 = *(const u32x4*)(base);
  u32x4 a1 = *(const u32x4*)(base + 16);
#pragma unroll
  for (int v = 0; v < 4; ++v) { a.u[v] = a0[v]; a.u[4 + v] = a1[v]; }
  return a;
}
// B-fragment: per-lane 16 bf16 = two contiguous 8-element runs (b128 each).
__device__ __forceinline__ Frag16 load_bfrag(const unsigned short* base) {
  Frag16 bf;
  u32x4 b0 = *(const u32x4*)(base);
  u32x4 b1 = *(const u32x4*)(base + 16);
#pragma unroll
  for (int v = 0; v < 4; ++v) { bf.u[v] = b0[v]; bf.u[4 + v] = b1[v]; }
  return bf;
}

// ---------------------------------------------------------------------------
// QKV GEMM (bf16 x bf16, f32 acc). Wave = 16(o) x 64(l); A reused over 4 WMMAs.
// ---------------------------------------------------------------------------
__global__ __launch_bounds__(256)
void qkv_gemm_kernel(const unsigned short* __restrict__ xt,
                     const unsigned short* __restrict__ wbf,
                     unsigned short* __restrict__ qw,
                     unsigned short* __restrict__ kw,
                     unsigned short* __restrict__ vw) {
  const int lane = threadIdx.x & 31;
  const int wave = threadIdx.x >> 5;
  const int b    = blockIdx.z;
  const int m0   = blockIdx.y * 16;
  const int n0   = (blockIdx.x * 8 + wave) * 64;

  const int row  = lane & 15;
  const int half = lane >> 4;
  const int kb0A = half * 8;
  const int kbB  = half * 8;
  const int col  = lane & 15;

  const unsigned short* arow = wbf + (size_t)(m0 + row) * C_ + kb0A;
  const unsigned short* xb   = xt + (size_t)b * L_ * C_ + kbB;

  v8f acc[4] = {};
  for (int k0 = 0; k0 < C_; k0 += 32) {
    Frag16 a = load_afrag(arow + k0);
#pragma unroll
    for (int t = 0; t < 4; ++t) {
      int l = n0 + t * 16 + col;
      Frag16 bf = load_bfrag(xb + (size_t)l * C_ + k0);
      acc[t] = __builtin_amdgcn_wmma_f32_16x16x32_bf16(false, a.v, false, bf.v,
                                                       (short)0, acc[t], false, false);
    }
  }

#pragma unroll
  for (int t = 0; t < 4; ++t)
#pragma unroll
    for (int r = 0; r < 8; ++r) {
      int o = m0 + r + half * 8;
      int l = n0 + t * 16 + col;
      float val = acc[t][r];
      if (o < HID_) {                                   // Q (pre-scaled)
        int h = o >> 6, d = o & 63;
        qw[((size_t)(b * H_ + h) * L_ + l) * DH_ + d] = f2bf(val * 0.125f);
      } else if (o < 2 * HID_) {                        // K
        int o2 = o - HID_; int h = o2 >> 6, d = o2 & 63;
        kw[((size_t)(b * H_ + h) * L_ + l) * DH_ + d] = f2bf(val);
      } else {                                          // V, d-major
        int o2 = o - 2 * HID_; int h = o2 >> 6, d = o2 & 63;
        vw[((size_t)(b * H_ + h) * DH_ + d) * L_ + l] = f2bf(val);
      }
    }
}

// ---------------------------------------------------------------------------
// Flash attention. Block = one (b,h); 8 waves = 8 query tiles of 16.
// K/V for each 32-key step staged once per block into LDS via async b128
// copies (double-buffered, ASYNCcnt + barriers). Online softmax per wave.
// LDS tile strides padded (72 / 40 ushorts) for conflict-free b128 reads.
// ---------------------------------------------------------------------------
#define KSTRIDE 72   // 64 data + 8 pad (16B-aligned rows, 36-dword stride)
#define VSTRIDE 40   // 32 data + 8 pad (16B-aligned rows, 20-dword stride)

__global__ __launch_bounds__(256)
void flash_attn_kernel(const unsigned short* __restrict__ qw,
                       const unsigned short* __restrict__ kw,
                       const unsigned short* __restrict__ vw,
                       unsigned short* __restrict__ ow) {
  __shared__ __align__(16) unsigned short kbuf[2][32 * KSTRIDE];
  __shared__ __align__(16) unsigned short vbuf[2][64 * VSTRIDE];
  __shared__ __align__(16) unsigned short pbuf[8][16 * 32];

  const int tid   = threadIdx.x;
  const int lane  = tid & 31;
  const int wave  = tid >> 5;
  const int bh    = blockIdx.x >> 4;          // 0..63
  const int tgrp  = blockIdx.x & 15;
  const int i0    = tgrp * 128 + wave * 16;

  const int row  = lane & 15;
  const int half = lane >> 4;
  const int kb0A = half * 8;
  const int kbB  = half * 8;
  const int col  = lane & 15;

  // staging assignments (256 threads, one b128 each for K and V per step)
  const int krow = tid >> 3, kchk = (tid & 7) * 8;    // K: 32 rows x 8 chunks of 8 ushorts
  const int vrow = tid >> 2, vchk = (tid & 3) * 8;    // V: 64 rows x 4 chunks of 8 ushorts
  const unsigned short* kg = kw + ((size_t)bh * L_ + krow) * DH_ + kchk;
  const unsigned short* vg = vw + ((size_t)bh * DH_ + vrow) * L_ + vchk;

  // Q: two K=32 A-fragments covering d=0..63 (b128 loads, loaded once)
  Frag16 aq[2];
  {
    const unsigned short* qr = qw + ((size_t)bh * L_ + (i0 + row)) * DH_ + kb0A;
    aq[0] = load_afrag(qr);
    aq[1] = load_afrag(qr + 32);
  }

  v8f oacc[4] = {};
  float rowmax[8], rowsum[8];
#pragma unroll
  for (int r = 0; r < 8; ++r) { rowmax[r] = -3.0e38f; rowsum[r] = 0.0f; }

  unsigned int* pu = (unsigned int*)&pbuf[wave][0];

  // prologue: stage j0 = 0 into buffer 0
  async_copy16(kg, &kbuf[0][krow * KSTRIDE + kchk]);
  async_copy16(vg, &vbuf[0][vrow * VSTRIDE + vchk]);

  int p = 0;
  for (int j0 = 0; j0 < L_; j0 += 32) {
    if (j0 + 32 < L_) {       // stage next step into the other buffer
      async_copy16(kg + (size_t)(j0 + 32) * DH_, &kbuf[1 - p][krow * KSTRIDE + kchk]);
      async_copy16(vg + (j0 + 32),               &vbuf[1 - p][vrow * VSTRIDE + vchk]);
      wait_async_le(2);       // oldest batch (current buffer) complete
    } else {
      wait_async_le(0);
    }
    __syncthreads();          // all waves' staged data visible

    // ---- S = Q K^T for two 16-column subtiles (fragments from LDS) ----
    v8f s[2];
#pragma unroll
    for (int jt = 0; jt < 2; ++jt) {
      const unsigned short* kr = &kbuf[p][(jt * 16 + col) * KSTRIDE + kbB];
      v8f sv = {};
#pragma unroll
      for (int c = 0; c < 2; ++c) {
        Frag16 bk = load_bfrag(kr + 32 * c);
        sv = __builtin_amdgcn_wmma_f32_16x16x32_bf16(false, aq[c].v, false, bk.v,
                                                     (short)0, sv, false, false);
      }
      s[jt] = sv;
    }

    // ---- online softmax (row reductions within 16-lane halves) ----
    float rmax[8];
#pragma unroll
    for (int r = 0; r < 8; ++r) rmax[r] = fmaxf(s[0][r], s[1][r]);
#pragma unroll
    for (int msk = 1; msk < 16; msk <<= 1)
#pragma unroll
      for (int r = 0; r < 8; ++r)
        rmax[r] = fmaxf(rmax[r], __shfl_xor(rmax[r], msk));

    float psum[8];
#pragma unroll
    for (int r = 0; r < 8; ++r) {
      float mn   = fmaxf(rowmax[r], rmax[r]);
      float corr = __expf(rowmax[r] - mn);
      rowmax[r]  = mn;
      float p0 = __expf(s[0][r] - mn);
      float p1 = __expf(s[1][r] - mn);
      s[0][r] = p0; s[1][r] = p1;
      psum[r] = p0 + p1;
      rowsum[r] *= corr;
#pragma unroll
      for (int t = 0; t < 4; ++t) oacc[t][r] *= corr;
    }
#pragma unroll
    for (int msk = 1; msk < 16; msk <<= 1)
#pragma unroll
      for (int r = 0; r < 8; ++r)
        psum[r] += __shfl_xor(psum[r], msk);
#pragma unroll
    for (int r = 0; r < 8; ++r) rowsum[r] += psum[r];

    // ---- P: C-layout -> per-wave LDS [m][j] -> A-layout ----
#pragma unroll
    for (int r = 0; r < 8; ++r) {
      int m = r + half * 8;
      pbuf[wave][m * 32 + col]      = f2bf(s[0][r]);
      pbuf[wave][m * 32 + 16 + col] = f2bf(s[1][r]);
    }
    asm volatile("s_wait_dscnt 0x0" ::: "memory");

    Frag16 pa;
#pragma unroll
    for (int v = 0; v < 8; ++v) {
      int K = ((v >> 2) << 4) + kb0A + ((v & 3) << 1);
      pa.u[v] = pu[(row * 32 + K) >> 1];
    }

    // ---- O += P V (V fragments from LDS, d-major rows) ----
#pragma unroll
    for (int t = 0; t < 4; ++t) {
      Frag16 bv = load_bfrag(&vbuf[p][(t * 16 + col) * VSTRIDE + kbB]);
      oacc[t] = __builtin_amdgcn_wmma_f32_16x16x32_bf16(false, pa.v, false, bv.v,
                                                        (short)0, oacc[t], false, false);
    }

    __syncthreads();          // buffer p fully consumed before it is re-staged
    p ^= 1;
  }

  // ---- epilogue: normalize, store [b, l, hid] bf16 ----
  const int b = bh >> 3, h = bh & 7;
#pragma unroll
  for (int t = 0; t < 4; ++t)
#pragma unroll
    for (int r = 0; r < 8; ++r) {
      int m   = r + half * 8;
      int l   = i0 + m;
      int hid = h * 64 + t * 16 + col;
      ow[((size_t)b * L_ + l) * HID_ + hid] = f2bf(oacc[t][r] / rowsum[r]);
    }
}

// ---------------------------------------------------------------------------
// Output projection GEMM + bias. Wave = 16(c) x 64(l).
// ---------------------------------------------------------------------------
__global__ __launch_bounds__(256)
void out_proj_kernel(const unsigned short* __restrict__ ow,
                     const unsigned short* __restrict__ wbf,
                     const float* __restrict__ bias,
                     float* __restrict__ out) {
  const int lane = threadIdx.x & 31;
  const int wave = threadIdx.x >> 5;
  const int b    = blockIdx.z;
  const int m0   = blockIdx.y * 16;
  const int n0   = (blockIdx.x * 8 + wave) * 64;
  const int row  = lane & 15, half = lane >> 4;
  const int kb0A = half * 8, kbB = half * 8, col = lane & 15;

  const unsigned short* arow = wbf + (size_t)(m0 + row) * HID_ + kb0A;
  const unsigned short* ob   = ow + (size_t)b * L_ * HID_ + kbB;

  v8f acc[4] = {};
  for (int k0 = 0; k0 < HID_; k0 += 32) {
    Frag16 a = load_afrag(arow + k0);
#pragma unroll
    for (int t = 0; t < 4; ++t) {
      int l = n0 + t * 16 + col;
      Frag16 bf = load_bfrag(ob + (size_t)l * HID_ + k0);
      acc[t] = __builtin_amdgcn_wmma_f32_16x16x32_bf16(false, a.v, false, bf.v,
                                                       (short)0, acc[t], false, false);
    }
  }
#pragma unroll
  for (int t = 0; t < 4; ++t)
#pragma unroll
    for (int r = 0; r < 8; ++r) {
      int c = m0 + r + half * 8;
      out[((size_t)b * C_ + c) * L_ + n0 + t * 16 + col] = acc[t][r] + bias[c];
    }
}

// ---------------------------------------------------------------------------
extern "C" void kernel_launch(void* const* d_in, const int* in_sizes, int n_in,
                              void* d_out, int out_size, void* d_ws, size_t ws_size,
                              hipStream_t stream) {
  (void)in_sizes; (void)n_in; (void)out_size; (void)ws_size;
  const float* x     = (const float*)d_in[0];
  const float* w_qkv = (const float*)d_in[1];
  const float* w_out = (const float*)d_in[2];
  const float* b_out = (const float*)d_in[3];
  float* out = (float*)d_out;

  unsigned short* ws = (unsigned short*)d_ws;
  const size_t NQ = (size_t)B_ * H_ * L_ * DH_;   // 8.39M bf16 elems (16 MB) each
  unsigned short* qw   = ws;
  unsigned short* kw   = ws + NQ;
  unsigned short* vw   = ws + 2 * NQ;
  unsigned short* ow   = ws + 3 * NQ;
  unsigned short* xt   = ws + 4 * NQ;              // [b,l,c] bf16
  unsigned short* wq_b = ws + 5 * NQ;              // bf16 w_qkv [1536,512]
  unsigned short* wo_b = wq_b + (size_t)3 * HID_ * C_;  // bf16 w_out [512,512]

  dim3 blk(256);
  cvt_w_kernel<<<dim3((3 * HID_ * C_ / 2 + 255) / 256), blk, 0, stream>>>(
      w_qkv, (unsigned int*)wq_b, 3 * HID_ * C_ / 2);
  cvt_w_kernel<<<dim3((C_ * HID_ / 2 + 255) / 256), blk, 0, stream>>>(
      w_out, (unsigned int*)wo_b, C_ * HID_ / 2);
  xpose_kernel<<<dim3(L_ / 32, C_ / 32, B_), blk, 0, stream>>>(x, xt);
  qkv_gemm_kernel<<<dim3(4, 96, 8), blk, 0, stream>>>(xt, wq_b, qw, kw, vw);
  flash_attn_kernel<<<dim3(1024), blk, 0, stream>>>(qw, kw, vw, ow);
  out_proj_kernel<<<dim3(4, 32, 8), blk, 0, stream>>>(ow, wo_b, b_out, out);
}
